// graph_construction_41609643164190
// MI455X (gfx1250) — matrix operand order repaired and checked by reading, not compiled, not run on hardware
//
#include <hip/hip_runtime.h>
#include <hip/hip_fp16.h>

typedef __attribute__((ext_vector_type(2)))  float    v2f;
typedef __attribute__((ext_vector_type(8)))  float    v8f;
typedef __attribute__((ext_vector_type(4)))  int      v4i;
typedef __attribute__((ext_vector_type(8)))  _Float16 v8h;
typedef __attribute__((ext_vector_type(16))) _Float16 v16h;

#define N_B   8
#define V_DIM 2048
#define C_DIM 64

// ---------------------------------------------------------------------------
// CDNA5 async global->LDS staging (ASYNCcnt-tracked), with safe fallback.
#if defined(__gfx1250__) && __has_builtin(__builtin_amdgcn_global_load_async_to_lds_b128)
#define HAS_ASYNC 1
__device__ __forceinline__ void async_cp16(const _Float16* g, _Float16* l) {
    __builtin_amdgcn_global_load_async_to_lds_b128(
        (v4i*)g,
        (__attribute__((address_space(3))) v4i*)l,
        0, 0);
}
__device__ __forceinline__ void wait_async() {
#if __has_builtin(__builtin_amdgcn_s_wait_asynccnt)
    __builtin_amdgcn_s_wait_asynccnt(0);
#else
    asm volatile("s_wait_asynccnt 0" ::: "memory");
#endif
}
#else
#define HAS_ASYNC 0
__device__ __forceinline__ void async_cp16(const _Float16* g, _Float16* l) {
    *(v8h*)l = *(const v8h*)g;
}
__device__ __forceinline__ void wait_async() {}
#endif

// ---------------------------------------------------------------------------
// weight f32 (V x V) -> f16 TRANSPOSED (Wt[n][k] = W[k][n]), tiled 32x32.
__global__ void k_transpose_weight(const float* __restrict__ w, _Float16* __restrict__ wt) {
    __shared__ _Float16 tile[32][33];
    const int t  = threadIdx.x;
    const int r  = t >> 3;          // 0..31
    const int c8 = (t & 7) * 4;     // 0,4,...,28
    const int bx = blockIdx.x * 32; // source col block
    const int by = blockIdx.y * 32; // source row block
    #pragma unroll
    for (int i = 0; i < 4; ++i)
        tile[r][c8 + i] = (_Float16)w[(size_t)(by + r) * V_DIM + bx + c8 + i];
    __syncthreads();
    #pragma unroll
    for (int i = 0; i < 4; ++i)
        wt[(size_t)(bx + r) * V_DIM + by + c8 + i] = tile[c8 + i][r];
}

// per-row sum of squares of x: sq[n*V + v]
__global__ void k_rowsq(const float* __restrict__ x, float* __restrict__ sq) {
    int r = blockIdx.x * 256 + threadIdx.x;     // 0 .. N*V-1
    const float* p = x + (size_t)r * C_DIM;
    float s = 0.f;
    #pragma unroll
    for (int c = 0; c < C_DIM; ++c) s += p[c] * p[c];
    sq[r] = s;
}

// ---------------------------------------------------------------------------
// adj[n,i,j] = exp(-0.5 * ||x_i - x_j||)  via f32 WMMA Gram matrix.
// One wave computes one 16x16 tile; block = 4 waves = 16x64 tile strip.
__global__ void k_adj(const float* __restrict__ x, const float* __restrict__ sq,
                      _Float16* __restrict__ adjh) {
    const int n    = blockIdx.z;
    const int i0   = blockIdx.y * 16;
    const int wave = threadIdx.x >> 5;
    const int lane = threadIdx.x & 31;
    const int j0   = blockIdx.x * 64 + wave * 16;
    const int lhi  = lane >> 4, llo = lane & 15;

    const float* xn   = x + (size_t)n * V_DIM * C_DIM;
    // A frag (16x4 f32): lane holds row M=llo, K = lhi*2 + {0,1}  -> float2 load
    const float* arow = xn + (size_t)(i0 + llo) * C_DIM + lhi * 2;
    // B frag (4x16 f32): lane holds col N=llo, K = lhi*2 + {0,1}; B = X^T so read row j0+llo
    const float* brow = xn + (size_t)(j0 + llo) * C_DIM + lhi * 2;

    v8f acc = {};
    #pragma unroll
    for (int kk = 0; kk < C_DIM; kk += 4) {
        v2f a = *(const v2f*)(arow + kk);
        v2f b = *(const v2f*)(brow + kk);
        acc = __builtin_amdgcn_wmma_f32_16x16x4_f32(
            false, a, false, b, (short)0, acc, false, false);
    }

    const float* sqn = sq + (size_t)n * V_DIM;
    float sqj = sqn[j0 + llo];
    // C/D layout: VGPR g -> row M = lhi*8 + g, col N = llo
    _Float16* out = adjh + ((size_t)n * V_DIM + i0 + lhi * 8) * V_DIM + (j0 + llo);
    #pragma unroll
    for (int g = 0; g < 8; ++g) {
        float d2 = sqn[i0 + lhi * 8 + g] + sqj - 2.0f * acc[g];
        d2 = d2 > 0.f ? d2 : 0.f;
        float a  = __expf(-0.5f * sqrtf(d2));
        out[(size_t)g * V_DIM] = (_Float16)a;
    }
}

// ---------------------------------------------------------------------------
// O[n] = adjh[n] (VxV f16) @ W (VxV f16, given transposed as Wt[n][k]).
// Block: 256 threads (8 waves), 128x128 macro-tile, BK=32,
// double-buffered LDS tiles staged with async global->LDS b128 copies.
#define BM 128
#define BN 128
#define BK 32
#define LDSS 40   // f16 row stride: 80B rows -> conflict-free b128 LDS reads

__global__ void __launch_bounds__(256, 2)
k_gemm(const _Float16* __restrict__ A, const _Float16* __restrict__ Wt,
       float* __restrict__ O) {
    __shared__ _Float16 As[2][BM * LDSS];   // [m][k]
    __shared__ _Float16 Bs[2][BN * LDSS];   // [n][k]   (Wt is already n-major)

    const int n    = blockIdx.z;
    const int m0   = blockIdx.y * BM;
    const int n0   = blockIdx.x * BN;
    const int t    = threadIdx.x;
    const int lane = t & 31;
    const int wv   = t >> 5;
    const int mw   = wv >> 1;           // 0..3 : wave row (32 rows each)
    const int nw   = wv & 1;            // 0..1 : wave col (64 cols each)
    const int lhi  = lane >> 4, llo = lane & 15;

    const _Float16* An = A + (size_t)n * V_DIM * V_DIM;

    // 512 chunks of 8 f16 per tile -> 2 per thread, all 16B-aligned b128 copies
    auto stageA = [&](int k0, int b) {
        #pragma unroll
        for (int c = 0; c < 2; ++c) {
            int ch = t + c * 256, row = ch >> 2, col = (ch & 3) * 8;
            async_cp16(An + (size_t)(m0 + row) * V_DIM + k0 + col,
                       &As[b][row * LDSS + col]);
        }
    };
    auto stageB = [&](int k0, int b) {
        #pragma unroll
        for (int c = 0; c < 2; ++c) {
            int ch = t + c * 256, row = ch >> 2, col = (ch & 3) * 8;
            async_cp16(Wt + (size_t)(n0 + row) * V_DIM + k0 + col,
                       &Bs[b][row * LDSS + col]);
        }
    };

    v8f acc[2][4] = {};
    stageA(0, 0); stageB(0, 0);

    int buf = 0;
    for (int k0 = 0; k0 < V_DIM; k0 += BK, buf ^= 1) {
        wait_async();        // this wave's async fills of As/Bs[buf] done
        __syncthreads();     // everyone's fills visible; prev compute finished
        if (k0 + BK < V_DIM) { stageA(k0 + BK, buf ^ 1); stageB(k0 + BK, buf ^ 1); }

        // A frag (16x32 f16): lane row M=llo, K = lhi*8 + {0..7}, then +16
        v16h af[2];
        #pragma unroll
        for (int mf = 0; mf < 2; ++mf) {
            int r = mw * 32 + mf * 16 + llo;
            v8h lo = *(const v8h*)&As[buf][r * LDSS + lhi * 8];
            v8h hi = *(const v8h*)&As[buf][r * LDSS + lhi * 8 + 16];
            af[mf] = __builtin_shufflevector(lo, hi,
                0,1,2,3,4,5,6,7,8,9,10,11,12,13,14,15);
        }
        #pragma unroll
        for (int nf = 0; nf < 4; ++nf) {
            // B frag (32x16 f16): lane col N=llo, K = lhi*8 + {0..7}, then +16
            int nn = nw * 64 + nf * 16 + llo;
            v8h lo = *(const v8h*)&Bs[buf][nn * LDSS + lhi * 8];
            v8h hi = *(const v8h*)&Bs[buf][nn * LDSS + lhi * 8 + 16];
            v16h bf = __builtin_shufflevector(lo, hi,
                0,1,2,3,4,5,6,7,8,9,10,11,12,13,14,15);
            #pragma unroll
            for (int mf = 0; mf < 2; ++mf)
                acc[mf][nf] = __builtin_amdgcn_wmma_f32_16x16x32_f16(
                    false, af[mf], false, bf, (short)0, acc[mf][nf], false, false);
        }
    }

    float* On = O + (size_t)n * V_DIM * V_DIM;
    #pragma unroll
    for (int mf = 0; mf < 2; ++mf)
        #pragma unroll
        for (int nf = 0; nf < 4; ++nf)
            #pragma unroll
            for (int g = 0; g < 8; ++g) {
                int r  = m0 + mw * 32 + mf * 16 + lhi * 8 + g;
                int cc = n0 + nw * 64 + nf * 16 + llo;
                On[(size_t)r * V_DIM + cc] = acc[mf][nf][g];
            }
}

// ---------------------------------------------------------------------------
// in-place row softmax, one block (256 thr) per row of 2048
__global__ void k_softmax(float* __restrict__ O) {
    __shared__ float red[256];
    const size_t base = (size_t)blockIdx.x * V_DIM;
    const int t = threadIdx.x;
    float v[8];
    float m = -1e30f;
    #pragma unroll
    for (int i = 0; i < 8; ++i) { v[i] = O[base + t + i * 256]; m = fmaxf(m, v[i]); }
    red[t] = m; __syncthreads();
    for (int s = 128; s > 0; s >>= 1) { if (t < s) red[t] = fmaxf(red[t], red[t + s]); __syncthreads(); }
    m = red[0]; __syncthreads();
    float sum = 0.f;
    #pragma unroll
    for (int i = 0; i < 8; ++i) { v[i] = __expf(v[i] - m); sum += v[i]; }
    red[t] = sum; __syncthreads();
    for (int s = 128; s > 0; s >>= 1) { if (t < s) red[t] += red[t + s]; __syncthreads(); }
    float inv = 1.0f / red[0];
    #pragma unroll
    for (int i = 0; i < 8; ++i) O[base + t + i * 256] = v[i] * inv;
}

// ---------------------------------------------------------------------------
extern "C" void kernel_launch(void* const* d_in, const int* in_sizes, int n_in,
                              void* d_out, int out_size, void* d_ws, size_t ws_size,
                              hipStream_t stream) {
    const float* x = (const float*)d_in[0];   // (N, V, C) f32
    const float* w = (const float*)d_in[1];   // (V, V)    f32
    float* out = (float*)d_out;               // (N, V, V) f32

    char* ws = (char*)d_ws;
    const size_t adj_bytes = (size_t)N_B * V_DIM * V_DIM * sizeof(_Float16); // 64 MB
    const size_t wt_bytes  = (size_t)V_DIM * V_DIM * sizeof(_Float16);       //  8 MB
    _Float16* adjh = (_Float16*)ws;
    _Float16* wt   = (_Float16*)(ws + adj_bytes);
    float*    sq   = (float*)(ws + adj_bytes + wt_bytes);

    dim3 gT(V_DIM / 32, V_DIM / 32);
    k_transpose_weight<<<gT, 256, 0, stream>>>(w, wt);
    k_rowsq<<<(N_B * V_DIM) / 256, 256, 0, stream>>>(x, sq);

    dim3 gA(V_DIM / 64, V_DIM / 16, N_B);
    k_adj<<<gA, 128, 0, stream>>>(x, sq, adjh);

    dim3 gG(V_DIM / BN, V_DIM / BM, N_B);
    k_gemm<<<gG, 256, 0, stream>>>(adjh, wt, out);

    k_softmax<<<N_B * V_DIM, 256, 0, stream>>>(out);
}
